// DiTMoEBlock_40742059770496
// MI455X (gfx1250) — matrix operand order repaired and checked by reading, not compiled, not run on hardware
//
#include <hip/hip_runtime.h>
#include <hip/hip_bf16.h>
#include <math.h>

// ---------------- problem constants (match reference) ----------------
static constexpr int B    = 4;
static constexpr int S    = 1024;
static constexpr int H    = 1024;
static constexpr int NH   = 16;
static constexpr int HD   = 64;
static constexpr int E    = 8;
static constexpr int I_   = 1024;
static constexpr int NSH  = 2;
static constexpr int NT   = B * S;      // 4096 tokens
static constexpr int SIXH = 6 * H;      // 6144
static constexpr float LNEPS = 1e-6f;

// ---------------- WMMA types (gfx1250, wave32) ----------------
typedef __attribute__((ext_vector_type(16))) __bf16 v16bf;
typedef __attribute__((ext_vector_type(8)))  float  v8f;

// ---------------- GEMM tile config ----------------
static constexpr int BM = 128, BN = 64, BK = 32;
static constexpr int LDA_S  = BK + 4;   // 36 floats: row stride of A tile in LDS
static constexpr int LDB_S  = BN + 4;   // 68 floats: row stride of B tile (non-trans)
static constexpr int LDBT_S = BK + 4;   // 36 floats: row stride of Bt tile (trans)
static constexpr int ATILE  = BM * LDA_S;            // 4608 floats per A buffer
static constexpr int BTILE  = BN * LDBT_S;           // 2304 floats (>= 32*68 too... see below)
// NOTE: non-trans B tile needs 32*68 = 2176 floats <= BTILE (2304). OK.

enum { EPI_STORE = 0, EPI_GELU = 1, EPI_RES_GATE = 2, EPI_MOE = 3, EPI_ACC = 4 };

__device__ __forceinline__ float gelu_f(float x) {
  return 0.5f * x * (1.0f + tanhf(0.7978845608028654f * (x + 0.044715f * x * x * x)));
}

// Async copy of 16 bytes global -> LDS (CDNA5 GLOBAL_LOAD_ASYNC_TO_LDS_B128,
// tracked by ASYNCcnt; bypasses VGPRs entirely).
__device__ __forceinline__ void async_ld16(unsigned lds_addr, const float* g) {
  asm volatile("global_load_async_to_lds_b128 %0, %1, off"
               :: "v"(lds_addr), "v"(g) : "memory");
}
// Wait until at most N of this wave's async copies are still in flight.
// Async global->LDS loads complete in order (ISA 10.4.1), so ASYNCcnt<=6 with
// two 6-copy tiles outstanding means the older tile has fully landed.
__device__ __forceinline__ void async_wait0() {
  asm volatile("s_wait_asynccnt 0x0" ::: "memory");
}
__device__ __forceinline__ void async_wait6() {
  asm volatile("s_wait_asynccnt 0x6" ::: "memory");
}

// C[M,N] (f32) = A[M,K] (f32, row-major) x B[K,N] (f32; TRANSB -> B given as [N,K])
// Double-buffered f32 LDS tiles filled by async global->LDS copies; bf16
// conversion at fragment-read time; f32 accumulate via v_wmma_f32_16x16x32_bf16.
// 256 threads = 8 waves; waves 4x2; each wave owns a 32x32 output tile (2x2 frags).
template <int EPI, bool TRANSB>
__global__ __launch_bounds__(256) void k_gemm(
    const float* __restrict__ A, int lda,
    const float* __restrict__ Bm, int ldb,
    float* __restrict__ C, int ldc,
    int K,
    const float* __restrict__ aux0,   // RES_GATE: residual ; MOE: per-token weight
    const float* __restrict__ aux1)   // RES_GATE: gate vector base (mods + 2H)
{
  __shared__ __align__(16) float Asf[2][ATILE];   // 2 x 18 KB
  __shared__ __align__(16) float Bsf[2][BTILE];   // 2 x 9.2 KB

  const int tid  = threadIdx.x;
  const int lane = tid & 31;
  const int wave = tid >> 5;
  const int wy   = wave >> 1;          // 0..3
  const int wx   = wave & 1;           // 0..1
  const int m0g  = blockIdx.y * BM;
  const int n0g  = blockIdx.x * BN;
  const int hi   = lane >> 4;          // half-wave select
  const int l15  = lane & 15;

  const unsigned ldsA = (unsigned)(uintptr_t)(&Asf[0][0]);
  const unsigned ldsB = (unsigned)(uintptr_t)(&Bsf[0][0]);

  // Stage K-tile `it` into LDS buffer `buf` (6 async copy instructions per wave).
  auto stage = [&](int it, int buf) {
    const int k0 = it * BK;
    const unsigned aBase = ldsA + (unsigned)(buf * ATILE) * 4u;
    const unsigned bBase = ldsB + (unsigned)(buf * BTILE) * 4u;
#pragma unroll
    for (int i = 0; i < 4; ++i) {
      const int idx = tid + i * 256;           // 0..1023 float4 granules
      const int r = idx >> 3;
      const int c = (idx & 7) << 2;
      async_ld16(aBase + (unsigned)(r * LDA_S + c) * 4u,
                 A + (size_t)(m0g + r) * lda + k0 + c);
    }
    if (TRANSB) {
      // store row-major [BN][BK] (rows of B^T); transpose happens at frag read
#pragma unroll
      for (int i = 0; i < 2; ++i) {
        const int idx = tid + i * 256;         // 0..511
        const int n = idx >> 3;                // 0..63
        const int d = (idx & 7) << 2;          // 0..28
        async_ld16(bBase + (unsigned)(n * LDBT_S + d) * 4u,
                   Bm + (size_t)(n0g + n) * ldb + k0 + d);
      }
    } else {
#pragma unroll
      for (int i = 0; i < 2; ++i) {
        const int idx = tid + i * 256;
        const int r = idx >> 4;                // 0..31
        const int c = (idx & 15) << 2;         // 0..60
        async_ld16(bBase + (unsigned)(r * LDB_S + c) * 4u,
                   Bm + (size_t)(k0 + r) * ldb + n0g + c);
      }
    }
  };

  const v8f vzero = {0.f, 0.f, 0.f, 0.f, 0.f, 0.f, 0.f, 0.f};
  v8f acc[2][2];
  acc[0][0] = vzero; acc[0][1] = vzero; acc[1][0] = vzero; acc[1][1] = vzero;

  const int nIter = K / BK;
  stage(0, 0);                                  // prologue: tile 0 in flight

  for (int it = 0; it < nIter; ++it) {
    const int buf = it & 1;
    if (it + 1 < nIter) {
      stage(it + 1, buf ^ 1);  // overlap: next tile streams while we compute
      async_wait6();           // tile `it` (older 6 copies) has landed
    } else {
      async_wait0();
    }
    __syncthreads();           // all waves' tile-`it` data visible in LDS

    const float* __restrict__ Ab = &Asf[buf][0];
    const float* __restrict__ Bb = &Bsf[buf][0];

    // ---- build bf16 fragments from f32 LDS tiles ----
    // A 16x32 bf16 layout (ISA 7.12.2): lanes 0-15 row m, VGPR v0..3 -> K {0..7}+8*hi,
    // v4..7 -> K {16..23}+8*hi.
    // B 32x16 bf16 layout: lane col n; same K mapping as A (v0..3 -> K {0..7}+8*hi, v4..7 -> K {16..23}+8*hi).
    v16bf af[2], bf[2];
#pragma unroll
    for (int t = 0; t < 2; ++t) {
      const int row = wy * 32 + t * 16 + l15;
#pragma unroll
      for (int v = 0; v < 4; ++v) {
        const int ka = 2 * v + 8 * hi;
        af[t][2 * v]     = (__bf16)Ab[row * LDA_S + ka];
        af[t][2 * v + 1] = (__bf16)Ab[row * LDA_S + ka + 1];
        const int kb = 16 + 2 * v + 8 * hi;
        af[t][2 * v + 8] = (__bf16)Ab[row * LDA_S + kb];
        af[t][2 * v + 9] = (__bf16)Ab[row * LDA_S + kb + 1];
      }
      const int col = wx * 32 + t * 16 + l15;
#pragma unroll
      for (int v = 0; v < 8; ++v) {
        const int kk = 2 * (v & 3) + 8 * hi + 16 * (v >> 2);
        if (TRANSB) {
          bf[t][2 * v]     = (__bf16)Bb[col * LDBT_S + kk];
          bf[t][2 * v + 1] = (__bf16)Bb[col * LDBT_S + kk + 1];
        } else {
          bf[t][2 * v]     = (__bf16)Bb[kk * LDB_S + col];
          bf[t][2 * v + 1] = (__bf16)Bb[(kk + 1) * LDB_S + col];
        }
      }
    }
#pragma unroll
    for (int i2 = 0; i2 < 2; ++i2)
#pragma unroll
      for (int j2 = 0; j2 < 2; ++j2)
        acc[i2][j2] = __builtin_amdgcn_wmma_f32_16x16x32_bf16(
            false, af[i2], false, bf[j2], (short)0, acc[i2][j2], false, false);

    __syncthreads();  // no wave still reads buf before it+2 overwrites it
  }

  // ---- epilogue (C/D layout: lane col = l15, VGPR v -> row v + 8*hi) ----
#pragma unroll
  for (int i2 = 0; i2 < 2; ++i2) {
#pragma unroll
    for (int j2 = 0; j2 < 2; ++j2) {
#pragma unroll
      for (int v = 0; v < 8; ++v) {
        const int m = m0g + wy * 32 + i2 * 16 + v + 8 * hi;
        const int n = n0g + wx * 32 + j2 * 16 + l15;
        const float val = acc[i2][j2][v];
        const size_t off = (size_t)m * ldc + n;
        if (EPI == EPI_STORE) {
          C[off] = val;
        } else if (EPI == EPI_GELU) {
          C[off] = gelu_f(val);
        } else if (EPI == EPI_RES_GATE) {
          const int bb = m >> 10;                       // token / S
          C[off] = aux0[off] + aux1[bb * SIXH + n] * val;
        } else if (EPI == EPI_MOE) {
          C[off] += aux0[m] * val;
        } else {                                        // EPI_ACC
          C[off] += val;
        }
      }
    }
  }
}

// ---------------- mods = silu(cond) @ adaLN_W  [B, 6H] ----------------
__global__ __launch_bounds__(256) void k_adaln(
    const float* __restrict__ cond, const float* __restrict__ W,
    float* __restrict__ mods)
{
  const int j = blockIdx.x * 256 + threadIdx.x;   // 0..6143
  const int b = blockIdx.y;
  float acc = 0.f;
  for (int h = 0; h < H; ++h) {
    const float c = cond[b * H + h];
    const float s = c / (1.0f + expf(-c));
    acc += s * W[(size_t)h * SIXH + j];
  }
  mods[b * SIXH + j] = acc;
}

// ---------------- LayerNorm + adaLN modulate (one block per token) ----------------
__global__ __launch_bounds__(256) void k_ln_mod(
    const float* __restrict__ x, const float* __restrict__ sc,
    const float* __restrict__ bi, const float* __restrict__ mods,
    int shiftOff, int scaleOff, float* __restrict__ out)
{
  const int t = blockIdx.x;
  const int b = t >> 10;
  const int tid = threadIdx.x;
  __shared__ float red[256];
  float lx[4];
  float s1 = 0.f;
#pragma unroll
  for (int i = 0; i < 4; ++i) {
    lx[i] = x[(size_t)t * H + tid + i * 256];
    s1 += lx[i];
  }
  red[tid] = s1; __syncthreads();
  for (int w = 128; w > 0; w >>= 1) { if (tid < w) red[tid] += red[tid + w]; __syncthreads(); }
  const float mean = red[0] * (1.0f / H);
  __syncthreads();
  float s2 = 0.f;
#pragma unroll
  for (int i = 0; i < 4; ++i) { const float d = lx[i] - mean; s2 += d * d; }
  red[tid] = s2; __syncthreads();
  for (int w = 128; w > 0; w >>= 1) { if (tid < w) red[tid] += red[tid + w]; __syncthreads(); }
  const float rstd = rsqrtf(red[0] * (1.0f / H) + LNEPS);
#pragma unroll
  for (int i = 0; i < 4; ++i) {
    const int n = tid + i * 256;
    const float ln = (lx[i] - mean) * rstd * sc[n] + bi[n];
    const float shm = mods[b * SIXH + shiftOff + n];
    const float scm = mods[b * SIXH + scaleOff + n];
    out[(size_t)t * H + n] = ln * (1.0f + scm) + shm;
  }
}

// ---------------- row softmax (scale pre-applied) ----------------
__global__ __launch_bounds__(256) void k_softmax_row(
    const float* __restrict__ Sc, float* __restrict__ P, float scale)
{
  const int r = blockIdx.x;
  const int tid = threadIdx.x;
  __shared__ float red[256];
  float lx[4];
  float mx = -3.4e38f;
#pragma unroll
  for (int i = 0; i < 4; ++i) {
    lx[i] = Sc[(size_t)r * S + tid + i * 256] * scale;
    mx = fmaxf(mx, lx[i]);
  }
  red[tid] = mx; __syncthreads();
  for (int w = 128; w > 0; w >>= 1) { if (tid < w) red[tid] = fmaxf(red[tid], red[tid + w]); __syncthreads(); }
  mx = red[0]; __syncthreads();
  float sum = 0.f;
#pragma unroll
  for (int i = 0; i < 4; ++i) { lx[i] = expf(lx[i] - mx); sum += lx[i]; }
  red[tid] = sum; __syncthreads();
  for (int w = 128; w > 0; w >>= 1) { if (tid < w) red[tid] += red[tid + w]; __syncthreads(); }
  const float inv = 1.0f / red[0];
#pragma unroll
  for (int i = 0; i < 4; ++i) P[(size_t)r * S + tid + i * 256] = lx[i] * inv;
}

// ---------------- MoE gate logits: [NT, E] ----------------
__global__ __launch_bounds__(256) void k_gate_logits(
    const float* __restrict__ x, const float* __restrict__ gk,
    float* __restrict__ logits)
{
  const int t = blockIdx.x;
  const int tid = threadIdx.x;
  float part[E];
#pragma unroll
  for (int e = 0; e < E; ++e) part[e] = 0.f;
  for (int i = tid; i < H; i += 256) {
    const float xv = x[(size_t)t * H + i];
#pragma unroll
    for (int e = 0; e < E; ++e) part[e] += xv * gk[(size_t)e * H + i];
  }
  __shared__ float red[256];
  for (int e = 0; e < E; ++e) {
    red[tid] = part[e]; __syncthreads();
    for (int w = 128; w > 0; w >>= 1) { if (tid < w) red[tid] += red[tid + w]; __syncthreads(); }
    if (tid == 0) logits[t * E + e] = red[0];
    __syncthreads();
  }
}

// ---------------- softmax over experts, greedy top-2, normalized weights ----------------
__global__ __launch_bounds__(256) void k_gate_topk(
    const float* __restrict__ logits, float* __restrict__ wtok)
{
  const int t = blockIdx.x * 256 + threadIdx.x;
  if (t >= NT) return;
  float l[E];
  float mx = -3.4e38f;
#pragma unroll
  for (int e = 0; e < E; ++e) { l[e] = logits[t * E + e]; mx = fmaxf(mx, l[e]); }
  float sum = 0.f;
#pragma unroll
  for (int e = 0; e < E; ++e) { l[e] = expf(l[e] - mx); sum += l[e]; }
  const float inv = 1.0f / sum;
#pragma unroll
  for (int e = 0; e < E; ++e) l[e] *= inv;
  int i1 = 0;
#pragma unroll
  for (int e = 1; e < E; ++e) if (l[e] > l[i1]) i1 = e;
  int i2 = (i1 == 0) ? 1 : 0;
#pragma unroll
  for (int e = 0; e < E; ++e) if (e != i1 && l[e] > l[i2]) i2 = e;
  const float norm = 1.0f / (l[i1] + l[i2] + 1e-20f);
#pragma unroll
  for (int e = 0; e < E; ++e)
    wtok[(size_t)e * NT + t] = (e == i1) ? l[i1] * norm : (e == i2) ? l[i2] * norm : 0.f;
}

__global__ __launch_bounds__(256) void k_zero(float* __restrict__ p, int n) {
  const int i = blockIdx.x * 256 + threadIdx.x;
  if (i < n) p[i] = 0.f;
}

// ---------------- out = h + gate_mlp * y ----------------
__global__ __launch_bounds__(256) void k_final(
    const float* __restrict__ h, const float* __restrict__ y,
    const float* __restrict__ mods, float* __restrict__ out)
{
  const int idx = blockIdx.x * 256 + threadIdx.x;  // < NT*H
  const int n = idx & (H - 1);
  const int b = idx >> 20;                          // / (S*H)
  out[idx] = h[idx] + mods[b * SIXH + 5 * H + n] * y[idx];
}

// ---------------- host orchestration ----------------
extern "C" void kernel_launch(void* const* d_in, const int* in_sizes, int n_in,
                              void* d_out, int out_size, void* d_ws, size_t ws_size,
                              hipStream_t stream) {
  (void)in_sizes; (void)n_in; (void)out_size; (void)ws_size;
  const float* hidden = (const float*)d_in[0];
  const float* cond   = (const float*)d_in[1];
  const float* adaW   = (const float*)d_in[2];
  const float* ln1s   = (const float*)d_in[3];
  const float* ln1b   = (const float*)d_in[4];
  const float* ln2s   = (const float*)d_in[5];
  const float* ln2b   = (const float*)d_in[6];
  const float* Wq     = (const float*)d_in[7];
  const float* Wk     = (const float*)d_in[8];
  const float* Wv     = (const float*)d_in[9];
  const float* Wo     = (const float*)d_in[10];
  const float* gk     = (const float*)d_in[11];
  const float* We1    = (const float*)d_in[12];
  const float* We2    = (const float*)d_in[13];
  const float* Ws1    = (const float*)d_in[14];
  const float* Ws2    = (const float*)d_in[15];
  float* out = (float*)d_out;

  char* ws = (char*)d_ws;
  size_t off = 0;
  auto alloc = [&](size_t bytes) -> float* {
    float* p = (float*)(ws + off);
    off += (bytes + 255) & ~size_t(255);
    return p;
  };
  const size_t ACT = (size_t)NT * H * sizeof(float);   // 16 MB
  float* mods   = alloc((size_t)B * SIXH * sizeof(float));
  float* n1     = alloc(ACT);                 // normed1; later reused as eh
  float* q      = alloc(ACT);                 // q..k later reused as sh (32 MB)
  float* kbuf   = alloc(ACT);
  float* v      = alloc(ACT);                 // later reused as y
  float* scores = alloc((size_t)S * S * sizeof(float));
  float* P      = alloc((size_t)S * S * sizeof(float));
  float* attn   = alloc(ACT);
  float* hbuf   = alloc(ACT);
  float* n2     = alloc(ACT);
  float* logits = alloc((size_t)NT * E * sizeof(float));
  float* wtok   = alloc((size_t)E * NT * sizeof(float));
  float* eh = n1;   // safe: normed1 consumed before MoE
  float* sh = q;    // safe: q/k consumed after attention; spans q+k = 32 MB
  float* y  = v;    // safe: v consumed after attention

  const dim3 blk(256);
  const dim3 gProj(H / BN, NT / BM);          // (16, 32)

  // adaLN: mods = silu(cond) @ adaLN_W
  k_adaln<<<dim3(SIXH / 256, B), blk, 0, stream>>>(cond, adaW, mods);
  // LN1 + modulate(shift_msa, scale_msa)
  k_ln_mod<<<dim3(NT), blk, 0, stream>>>(hidden, ln1s, ln1b, mods, 0 * H, 1 * H, n1);

  // QKV projections (WMMA bf16)
  k_gemm<EPI_STORE, false><<<gProj, blk, 0, stream>>>(n1, H, Wq, H, q,    H, H, nullptr, nullptr);
  k_gemm<EPI_STORE, false><<<gProj, blk, 0, stream>>>(n1, H, Wk, H, kbuf, H, H, nullptr, nullptr);
  k_gemm<EPI_STORE, false><<<gProj, blk, 0, stream>>>(n1, H, Wv, H, v,    H, H, nullptr, nullptr);

  // attention per (batch, head): QK^T -> softmax -> PV
  const float iscale = 0.125f;                // 1/sqrt(HD)
  for (int b = 0; b < B; ++b) {
    for (int nh = 0; nh < NH; ++nh) {
      const float* Qb = q    + (size_t)b * S * H + nh * HD;
      const float* Kb = kbuf + (size_t)b * S * H + nh * HD;
      const float* Vb = v    + (size_t)b * S * H + nh * HD;
      float*       Ob = attn + (size_t)b * S * H + nh * HD;
      k_gemm<EPI_STORE, true ><<<dim3(S / BN, S / BM), blk, 0, stream>>>(
          Qb, H, Kb, H, scores, S, HD, nullptr, nullptr);
      k_softmax_row<<<dim3(S), blk, 0, stream>>>(scores, P, iscale);
      k_gemm<EPI_STORE, false><<<dim3(HD / BN, S / BM), blk, 0, stream>>>(
          P, S, Vb, H, Ob, H, S, nullptr, nullptr);
    }
  }

  // h = hidden + gate_msa * (attn @ Wo)
  k_gemm<EPI_RES_GATE, false><<<gProj, blk, 0, stream>>>(
      attn, H, Wo, H, hbuf, H, H, hidden, mods + 2 * H);

  // LN2 + modulate(shift_mlp, scale_mlp)
  k_ln_mod<<<dim3(NT), blk, 0, stream>>>(hbuf, ln2s, ln2b, mods, 3 * H, 4 * H, n2);

  // MoE gate
  k_gate_logits<<<dim3(NT), blk, 0, stream>>>(n2, gk, logits);
  k_gate_topk<<<dim3((NT + 255) / 256), blk, 0, stream>>>(logits, wtok);

  // y = 0, then routed experts: y += wtok[e] * (gelu(n2 @ We1[e]) @ We2[e])
  k_zero<<<dim3((NT * H + 255) / 256), blk, 0, stream>>>(y, NT * H);
  for (int e = 0; e < E; ++e) {
    const float* W1 = We1 + (size_t)e * H * I_;
    const float* W2 = We2 + (size_t)e * I_ * H;
    k_gemm<EPI_GELU, false><<<dim3(I_ / BN, NT / BM), blk, 0, stream>>>(
        n2, H, W1, I_, eh, I_, H, nullptr, nullptr);
    k_gemm<EPI_MOE, false><<<gProj, blk, 0, stream>>>(
        eh, I_, W2, H, y, H, I_, wtok + (size_t)e * NT, nullptr);
  }

  // shared experts: y += gelu(n2 @ Ws1) @ Ws2
  const int SI = NSH * I_;                    // 2048
  k_gemm<EPI_GELU, false><<<dim3(SI / BN, NT / BM), blk, 0, stream>>>(
      n2, H, Ws1, SI, sh, SI, H, nullptr, nullptr);
  k_gemm<EPI_ACC, false><<<gProj, blk, 0, stream>>>(
      sh, SI, Ws2, H, y, H, SI, nullptr, nullptr);

  // out = h + gate_mlp * y
  k_final<<<dim3(NT * H / 256), blk, 0, stream>>>(hbuf, y, mods, out);
}
